// keops_RBFkernel_88734024335763
// MI455X (gfx1250) — matrix operand order repaired and checked by reading, not compiled, not run on hardware
//
#include <hip/hip_runtime.h>

typedef __attribute__((ext_vector_type(2)))  float     v2f;
typedef __attribute__((ext_vector_type(8)))  float     v8f;
typedef __attribute__((ext_vector_type(2)))  _Float16  v2h;
typedef __attribute__((ext_vector_type(8)))  _Float16  v8h;
typedef __attribute__((ext_vector_type(16))) _Float16  v16h;

#define D_FEAT 32
#define D_AUG  36      // 32 features + [norm, const] + 2 zero pad (9 K-chunks of 4)
#define NCH    9       // K-chunks per distance GEMM
#define DV     16

// ---------------------------------------------------------------------------
// Pre-pass: build augmented, pre-scaled matrices so the distance GEMM computes
//   xa_i . ya_j = -g*log2(e) * ||x_i - y_j||^2   directly:
//     xa = [2*c*x_i, -c*||x_i||^2, -c, 0, 0],  ya = [y_j, 1, ||y_j||^2, 0, 0]
// with c = g*log2(e).  Also f16 transposed b (b16T[dv][j]) for the 2nd GEMM.
// ---------------------------------------------------------------------------
__global__ __launch_bounds__(256)
void rbf_prep(const float* __restrict__ ls, const float* __restrict__ x,
              const float* __restrict__ y, const float* __restrict__ b,
              float* __restrict__ xa, float* __restrict__ ya,
              _Float16* __restrict__ b16t, int N, int M) {
    const int i = blockIdx.x * blockDim.x + threadIdx.x;
    const float c = ls[0] * 1.44269504088896340736f;   // g * log2(e)

    if (i < N) {
        float s = 0.f;
#pragma unroll
        for (int d = 0; d < D_FEAT; ++d) {
            float v = x[(size_t)i * D_FEAT + d];
            s += v * v;
            xa[(size_t)i * D_AUG + d] = 2.f * c * v;
        }
        xa[(size_t)i * D_AUG + 32] = -c * s;
        xa[(size_t)i * D_AUG + 33] = -c;
        xa[(size_t)i * D_AUG + 34] = 0.f;
        xa[(size_t)i * D_AUG + 35] = 0.f;
    }
    if (i < M) {
        float s = 0.f;
#pragma unroll
        for (int d = 0; d < D_FEAT; ++d) {
            float v = y[(size_t)i * D_FEAT + d];
            s += v * v;
            ya[(size_t)i * D_AUG + d] = v;
        }
        ya[(size_t)i * D_AUG + 32] = 1.f;
        ya[(size_t)i * D_AUG + 33] = s;
        ya[(size_t)i * D_AUG + 34] = 0.f;
        ya[(size_t)i * D_AUG + 35] = 0.f;
#pragma unroll
        for (int v = 0; v < DV; ++v)
            b16t[(size_t)v * M + i] = (_Float16)b[(size_t)i * DV + v];
    }
}

// ---------------------------------------------------------------------------
// Per-jt register working set: 18 ya fragments (32 j's) + one b16T fragment.
// Double-buffered across jt iterations for software pipelining.  Prefetch
// indices are NOT clamped: the final (never-consumed) prefetch over-reads
// into the b16t/pad region of the workspace, keeping addresses as simple
// loop-carried increments.
// ---------------------------------------------------------------------------
struct JBuf {
    v2f  yf[2 * NCH];
    v16h bb;
};

__device__ __forceinline__
void load_jbuf(JBuf& bf, const float* __restrict__ ya,
               const _Float16* __restrict__ b16t,
               int jt, int ln, int hi, int M) {
#pragma unroll
    for (int t = 0; t < 2; ++t)
#pragma unroll
        for (int c = 0; c < NCH; ++c)
            bf.yf[t * NCH + c] =
                *(const v2f*)(ya + (size_t)(jt * 32 + t * 16 + ln) * D_AUG + 4 * c + 2 * hi);
    bf.bb = *(const v16h*)(b16t + (size_t)ln * M + jt * 32 + 16 * hi);
}

__device__ __forceinline__
void compute_jbuf(const JBuf& bf, v8f& acc, const v2f* afrag,
                  _Float16* ldsE, int lbase, int ln, int hi) {
#pragma unroll
    for (int t = 0; t < 2; ++t) {
        // dot = -g*log2(e) * sq   (exact fp32 WMMA chain over 9 K-chunks)
        v8f s = {0.f, 0.f, 0.f, 0.f, 0.f, 0.f, 0.f, 0.f};
#pragma unroll
        for (int c = 0; c < NCH; ++c)
            s = __builtin_amdgcn_wmma_f32_16x16x4_f32(
                    false, afrag[c], false, bf.yf[t * NCH + c], (short)0, s, false, false);

        // e = exp(-g*sq) = exp2(dot); dot <= 0 up to rounding noise, so no
        // clamp is needed (exp2(+eps) ~ 1 matches the clamped reference).
        // Pack pairs with v_cvt_pkrtz_f16_f32; odd element uses d16_hi store.
#pragma unroll
        for (int r = 0; r < 8; r += 2) {
            v2h p = __builtin_bit_cast(v2h,
                        __builtin_amdgcn_cvt_pkrtz(
                            __builtin_amdgcn_exp2f(s[r]),
                            __builtin_amdgcn_exp2f(s[r + 1])));
            ldsE[lbase + (r     + 8 * hi) * 32 + t * 16 + ln] = p[0];
            ldsE[lbase + (r + 1 + 8 * hi) * 32 + t * 16 + ln] = p[1];
        }
    }

    // order the transposing LDS loads after the stores above
    asm volatile("s_wait_dscnt 0" ::: "memory");

    // A fragment of E (f16 16x32 layout): row = ln,
    // halves 0..15 = K = 16*hi + 0..15
    union { v16h v; v8h h[2]; } ea;
    ea.h[0] = *(const v8h*)&ldsE[lbase + ln * 32 + 16 * hi];
    ea.h[1] = *(const v8h*)&ldsE[lbase + ln * 32 + 16 * hi + 8];

    acc = __builtin_amdgcn_wmma_f32_16x16x32_f16(
              false, ea.v, false, bf.bb, (short)0, acc, false, false);
}

// ---------------------------------------------------------------------------
// Main kernel: one wave per 16-row i-tile, jt loop unrolled x2 with register
// double buffering (loads for jt+1 issue before WMMAs of jt) and LDS tile
// double buffering (disjoint halves kill the cross-iteration WAR hazard).
// ---------------------------------------------------------------------------
__global__ __launch_bounds__(32)
void rbf_main(const float* __restrict__ xa, const float* __restrict__ ya,
              const _Float16* __restrict__ b16t,
              float* __restrict__ out, int N, int M) {
    __shared__ _Float16 ldsE[2 * 16 * 32];   // two 1 KB E tiles

    const int lane = threadIdx.x & 31;
    const int hi   = lane >> 4;
    const int ln   = lane & 15;
    const int i0   = blockIdx.x * 16;

    // A fragments of the augmented X tile: row = ln, K = 4c + 2*hi + {0,1}
    v2f afrag[NCH];
#pragma unroll
    for (int c = 0; c < NCH; ++c)
        afrag[c] = *(const v2f*)(xa + (size_t)(i0 + ln) * D_AUG + 4 * c + 2 * hi);

    v8f acc = {0.f, 0.f, 0.f, 0.f, 0.f, 0.f, 0.f, 0.f};

    const int njt = M / 32;        // 512, even
    JBuf bufA, bufB;
    load_jbuf(bufA, ya, b16t, 0, ln, hi, M);

    for (int jt = 0; jt < njt; jt += 2) {
        load_jbuf(bufB, ya, b16t, jt + 1, ln, hi, M);       // prefetch jt+1
        compute_jbuf(bufA, acc, afrag, ldsE, 0,   ln, hi);  // consume  jt
        load_jbuf(bufA, ya, b16t, jt + 2, ln, hi, M);       // prefetch jt+2 (tail over-reads pad)
        compute_jbuf(bufB, acc, afrag, ldsE, 512, ln, hi);  // consume  jt+1
    }

    // C-layout epilogue: element (m = r + 8*hi, n = ln)
#pragma unroll
    for (int r = 0; r < 8; ++r)
        out[(size_t)(i0 + r + 8 * hi) * DV + ln] = acc[r];
}

// ---------------------------------------------------------------------------
extern "C" void kernel_launch(void* const* d_in, const int* in_sizes, int n_in,
                              void* d_out, int out_size, void* d_ws, size_t ws_size,
                              hipStream_t stream) {
    const float* ls = (const float*)d_in[0];
    const float* x  = (const float*)d_in[1];
    const float* y  = (const float*)d_in[2];
    const float* b  = (const float*)d_in[3];
    float*       out = (float*)d_out;

    const int N = in_sizes[1] / D_FEAT;   // 16384
    const int M = in_sizes[2] / D_FEAT;   // 16384

    // ws layout: xa | ya | b16t | pad (absorbs tail prefetch over-read)
    float*    xa   = (float*)d_ws;                 // N x 36 f32
    float*    ya   = xa + (size_t)N * D_AUG;       // M x 36 f32
    _Float16* b16t = (_Float16*)(ya + (size_t)M * D_AUG);   // DV x M f16

    const int mx = (N > M) ? N : M;
    rbf_prep<<<(mx + 255) / 256, 256, 0, stream>>>(ls, x, y, b, xa, ya, b16t, N, M);
    rbf_main<<<N / 16, 32, 0, stream>>>(xa, ya, b16t, out, N, M);
}
